// MessagePassingLayer_57518202028476
// MI455X (gfx1250) — matrix operand order repaired and checked by reading, not compiled, not run on hardware
//
#include <hip/hip_runtime.h>

typedef _Float16 half_t;
typedef __attribute__((ext_vector_type(4)))  _Float16 v4h;
typedef __attribute__((ext_vector_type(8)))  _Float16 v8h;
typedef __attribute__((ext_vector_type(16))) _Float16 v16h;
typedef __attribute__((ext_vector_type(8)))  float    v8f;

#define N_NODES_C 100000
#define N_EDGES_C 1600000

// workspace layout (half elements)
#define MSG_W0_OFF 0        // [64][192] transposed
#define MSG_W1_OFF 12288    // 4 x [64][64]
#define UPD_W0_OFF 28672    // [64][128]
#define UPD_W1_OFF 36864    // 4 x [64][64]
#define W_TOTAL_HALF 53248
#define COUNTS_BYTE_OFF (W_TOTAL_HALF * 2)   // 106496 (32B aligned)

struct Biases { const float* b[5]; };

__global__ void zero_f32(float* __restrict__ p, int n) {
    int i = blockIdx.x * blockDim.x + threadIdx.x;
    if (i < n) p[i] = 0.0f;
}

// Wt[n*K + k] = (half) W[k*N + n]   (N is always 64 here)
__global__ void conv_w(const float* __restrict__ W, half_t* __restrict__ Wt, int K, int N) {
    int i = blockIdx.x * blockDim.x + threadIdx.x;
    if (i >= K * N) return;
    int k = i % K, n = i / K;
    Wt[n * K + k] = (half_t)W[k * N + n];
}

__device__ __forceinline__ v4h cvt4(float4 f) {
    v4h h;
    h[0] = (half_t)f.x; h[1] = (half_t)f.y; h[2] = (half_t)f.z; h[3] = (half_t)f.w;
    return h;
}

// A fragment: 16x32 fp16 tile from row-major LDS [16][lda], K-window at kbase.
// contiguous K order (same as gB): lane m = lane&15; lanes0-15 hold K=kbase..+15,
// lanes16-31 hold K=kbase+16..+31.
__device__ __forceinline__ v16h ldsA(const half_t* base, int lda, int kbase, int lane) {
    int m  = lane & 15;
    int kh = (lane >> 4) << 4;
    const half_t* p = base + m * lda + kbase + kh;
    v16h a;
    ((v8h*)&a)[0] = *(const v8h*)(p);
    ((v8h*)&a)[1] = *(const v8h*)(p + 8);
    return a;
}

// B fragment: 32x16 fp16 from global transposed weights Wt[n*K + k].
// lane column n = nbase + (lane&15); lanes0-15 K=kbase..+15, lanes16-31 K=kbase+16..+31.
__device__ __forceinline__ v16h gB(const half_t* __restrict__ Wt, int K, int nbase, int kbase, int lane) {
    const half_t* p = Wt + (nbase + (lane & 15)) * K + kbase + ((lane >> 4) << 4);
    return *(const v16h*)p;
}

__global__ __launch_bounds__(128)
void msg_kernel(const float* __restrict__ h_v,
                const int*   __restrict__ src,
                const int*   __restrict__ dst,
                const float* __restrict__ h_e,
                const half_t* __restrict__ wbase,
                Biases bi,
                float* __restrict__ summed,
                float* __restrict__ counts) {
    __shared__ alignas(16) half_t in0[16 * 192];
    __shared__ alignas(16) half_t act0[16 * 64];
    __shared__ alignas(16) half_t act1[16 * 64];
    __shared__ int ssrc[16];
    __shared__ int sdst[16];

    const int tid   = threadIdx.x;
    const int lane  = tid & 31;
    const int wave  = tid >> 5;
    const int ebase = blockIdx.x * 16;

    // cache the 16 edge endpoints once; bump counts
    if (tid < 16) {
        int s = src[ebase + tid];
        int d = dst[ebase + tid];
        ssrc[tid] = s;
        sdst[tid] = d;
        atomicAdd(&counts[d], 1.0f);
    }
    __syncthreads();

    // ---- gather [16 x 192] input as 3 segments of [16 x 64] floats,
    //      float4 (b128) loads + 8B packed half stores, power-of-2 indexing ----
    #pragma unroll
    for (int j = 0; j < 2; ++j) {
        int idx = tid + j * 128;           // 0..255 -> 16 rows x 16 float4
        int row = idx >> 4;
        int q   = (idx & 15) << 2;         // float column within segment
        float4 a = *(const float4*)(h_v + (long)ssrc[row] * 64 + q);
        float4 b = *(const float4*)(h_v + (long)sdst[row] * 64 + q);
        float4 c = *(const float4*)(h_e + (long)(ebase + row) * 64 + q);
        half_t* rp = in0 + row * 192;
        *(v4h*)(rp + q)       = cvt4(a);
        *(v4h*)(rp + 64 + q)  = cvt4(b);
        *(v4h*)(rp + 128 + q) = cvt4(c);
    }
    __syncthreads();

    const int nbase = wave * 16;          // each wave owns 16 output columns
    const int nl    = lane & 15;
    const int mb    = (lane >> 4) * 8;    // accumulator row base per lane-half

    // ---- layer 0: [16x192] @ [192x64], K = 6 x 32 ----
    v8f c = {};
    const half_t* W0 = wbase + MSG_W0_OFF;
    #pragma unroll
    for (int kt = 0; kt < 6; ++kt) {
        v16h a = ldsA(in0, 192, kt * 32, lane);
        v16h b = gB(W0, 192, nbase, kt * 32, lane);
        c = __builtin_amdgcn_wmma_f32_16x16x32_f16(false, a, false, b, (short)0, c, false, false);
    }
    {
        float bb = bi.b[0][nbase + nl];
        #pragma unroll
        for (int i = 0; i < 8; ++i)
            act0[(mb + i) * 64 + nbase + nl] = (half_t)fmaxf(c[i] + bb, 0.0f);
    }
    __syncthreads();

    // ---- hidden layers 1..3 (ReLU), double-buffered ----
    const half_t* Wh[4] = { wbase + MSG_W1_OFF,        wbase + MSG_W1_OFF + 4096,
                            wbase + MSG_W1_OFF + 8192, wbase + MSG_W1_OFF + 12288 };
    half_t* bufs[2] = { act0, act1 };
    #pragma unroll
    for (int L = 0; L < 3; ++L) {
        const half_t* aIn  = bufs[L & 1];
        half_t*       aOut = bufs[(L + 1) & 1];
        v8f cc = {};
        #pragma unroll
        for (int kt = 0; kt < 2; ++kt) {
            v16h a = ldsA(aIn, 64, kt * 32, lane);
            v16h b = gB(Wh[L], 64, nbase, kt * 32, lane);
            cc = __builtin_amdgcn_wmma_f32_16x16x32_f16(false, a, false, b, (short)0, cc, false, false);
        }
        float bb = bi.b[L + 1][nbase + nl];
        #pragma unroll
        for (int i = 0; i < 8; ++i)
            aOut[(mb + i) * 64 + nbase + nl] = (half_t)fmaxf(cc[i] + bb, 0.0f);
        __syncthreads();
    }

    // ---- final layer (no ReLU) + scatter-add by dst ----
    {
        const half_t* aIn = bufs[1];      // L=2 wrote act1
        v8f cc = {};
        #pragma unroll
        for (int kt = 0; kt < 2; ++kt) {
            v16h a = ldsA(aIn, 64, kt * 32, lane);
            v16h b = gB(Wh[3], 64, nbase, kt * 32, lane);
            cc = __builtin_amdgcn_wmma_f32_16x16x32_f16(false, a, false, b, (short)0, cc, false, false);
        }
        float bb = bi.b[4][nbase + nl];
        #pragma unroll
        for (int i = 0; i < 8; ++i) {
            int m = mb + i;
            atomicAdd(&summed[(long)sdst[m] * 64 + nbase + nl], cc[i] + bb);
        }
    }
}

__global__ __launch_bounds__(128)
void upd_kernel(const float* __restrict__ h_v,
                const half_t* __restrict__ wbase,
                Biases bi,
                const float* __restrict__ counts,
                float* __restrict__ out) {   // out holds `summed` on entry; overwritten in place
    __shared__ alignas(16) half_t in0[16 * 128];
    __shared__ alignas(16) half_t act0[16 * 64];
    __shared__ alignas(16) half_t act1[16 * 64];

    const int tid   = threadIdx.x;
    const int lane  = tid & 31;
    const int wave  = tid >> 5;
    const int vbase = blockIdx.x * 16;

    // ---- build [16 x 128] input: [h_v | summed/clamp(count)] via float4 loads ----
    #pragma unroll
    for (int j = 0; j < 2; ++j) {
        int idx = tid + j * 128;           // 0..255 -> 16 rows x 16 float4
        int row = idx >> 4;
        int q   = (idx & 15) << 2;
        long node = vbase + row;
        float4 a = *(const float4*)(h_v + node * 64 + q);
        float4 s = *(const float4*)(out + node * 64 + q);
        float inv = 1.0f / fmaxf(counts[node], 1.0f);
        s.x *= inv; s.y *= inv; s.z *= inv; s.w *= inv;
        half_t* rp = in0 + row * 128;
        *(v4h*)(rp + q)      = cvt4(a);
        *(v4h*)(rp + 64 + q) = cvt4(s);
    }
    __syncthreads();

    const int nbase = wave * 16;
    const int nl    = lane & 15;
    const int mb    = (lane >> 4) * 8;

    // ---- layer 0: K = 128 = 4 x 32 ----
    v8f c = {};
    const half_t* W0 = wbase + UPD_W0_OFF;
    #pragma unroll
    for (int kt = 0; kt < 4; ++kt) {
        v16h a = ldsA(in0, 128, kt * 32, lane);
        v16h b = gB(W0, 128, nbase, kt * 32, lane);
        c = __builtin_amdgcn_wmma_f32_16x16x32_f16(false, a, false, b, (short)0, c, false, false);
    }
    {
        float bb = bi.b[0][nbase + nl];
        #pragma unroll
        for (int i = 0; i < 8; ++i)
            act0[(mb + i) * 64 + nbase + nl] = (half_t)fmaxf(c[i] + bb, 0.0f);
    }
    __syncthreads();

    const half_t* Wh[4] = { wbase + UPD_W1_OFF,        wbase + UPD_W1_OFF + 4096,
                            wbase + UPD_W1_OFF + 8192, wbase + UPD_W1_OFF + 12288 };
    half_t* bufs[2] = { act0, act1 };
    #pragma unroll
    for (int L = 0; L < 3; ++L) {
        const half_t* aIn  = bufs[L & 1];
        half_t*       aOut = bufs[(L + 1) & 1];
        v8f cc = {};
        #pragma unroll
        for (int kt = 0; kt < 2; ++kt) {
            v16h a = ldsA(aIn, 64, kt * 32, lane);
            v16h b = gB(Wh[L], 64, nbase, kt * 32, lane);
            cc = __builtin_amdgcn_wmma_f32_16x16x32_f16(false, a, false, b, (short)0, cc, false, false);
        }
        float bb = bi.b[L + 1][nbase + nl];
        #pragma unroll
        for (int i = 0; i < 8; ++i)
            aOut[(mb + i) * 64 + nbase + nl] = (half_t)fmaxf(cc[i] + bb, 0.0f);
        __syncthreads();
    }

    // ---- final layer (no ReLU): write result in place ----
    {
        const half_t* aIn = bufs[1];
        v8f cc = {};
        #pragma unroll
        for (int kt = 0; kt < 2; ++kt) {
            v16h a = ldsA(aIn, 64, kt * 32, lane);
            v16h b = gB(Wh[3], 64, nbase, kt * 32, lane);
            cc = __builtin_amdgcn_wmma_f32_16x16x32_f16(false, a, false, b, (short)0, cc, false, false);
        }
        float bb = bi.b[4][nbase + nl];
        #pragma unroll
        for (int i = 0; i < 8; ++i) {
            long node = vbase + mb + i;
            out[node * 64 + nbase + nl] = cc[i] + bb;
        }
    }
}

extern "C" void kernel_launch(void* const* d_in, const int* in_sizes, int n_in,
                              void* d_out, int out_size, void* d_ws, size_t ws_size,
                              hipStream_t stream) {
    const float* h_v = (const float*)d_in[0];
    const int*   ei  = (const int*)d_in[1];     // [2, E] int32
    const float* h_e = (const float*)d_in[2];

    const float *mW[5], *uW[5];
    Biases mB, uB;
    for (int i = 0; i < 5; ++i) {
        mW[i]   = (const float*)d_in[3 + 2 * i];
        mB.b[i] = (const float*)d_in[4 + 2 * i];
        uW[i]   = (const float*)d_in[13 + 2 * i];
        uB.b[i] = (const float*)d_in[14 + 2 * i];
    }

    half_t* wbase  = (half_t*)d_ws;
    float*  counts = (float*)((char*)d_ws + COUNTS_BYTE_OFF);
    float*  out    = (float*)d_out;

    // zero the scatter buffer (d_out doubles as `summed`) and counts
    zero_f32<<<(N_NODES_C * 64 + 255) / 256, 256, 0, stream>>>(out, N_NODES_C * 64);
    zero_f32<<<(N_NODES_C + 255) / 256, 256, 0, stream>>>(counts, N_NODES_C);

    // convert + transpose weights to fp16
    conv_w<<<(192 * 64 + 255) / 256, 256, 0, stream>>>(mW[0], wbase + MSG_W0_OFF, 192, 64);
    for (int i = 1; i < 5; ++i)
        conv_w<<<(64 * 64 + 255) / 256, 256, 0, stream>>>(mW[i], wbase + MSG_W1_OFF + (i - 1) * 4096, 64, 64);
    conv_w<<<(128 * 64 + 255) / 256, 256, 0, stream>>>(uW[0], wbase + UPD_W0_OFF, 128, 64);
    for (int i = 1; i < 5; ++i)
        conv_w<<<(64 * 64 + 255) / 256, 256, 0, stream>>>(uW[i], wbase + UPD_W1_OFF + (i - 1) * 4096, 64, 64);

    // per-edge message MLP + scatter-add (E is an exact multiple of 16)
    msg_kernel<<<N_EDGES_C / 16, 128, 0, stream>>>(h_v, ei, ei + N_EDGES_C, h_e,
                                                   wbase, mB, out, counts);

    // per-node update MLP (N is an exact multiple of 16), in-place on d_out
    upd_kernel<<<N_NODES_C / 16, 128, 0, stream>>>(h_v, wbase, uB, counts, out);
}
